// Encoder_41137196761551
// MI455X (gfx1250) — compile-verified, matrix operand order checked by reading, not run on hardware
//
#include <hip/hip_runtime.h>

// ---------------------------------------------------------------------------
// 2-layer transformer encoder, B=8 S=1024 D=1024 H=16 DH=64 DFF=4096, fp32 I/O.
// GEMM-shaped math via v_wmma_f32_16x16x32_bf16 (wave32 CDNA5).
// bf16 operands in global memory; GEMM A/B tiles and flash Q/K tiles stream
// through the Tensor Data Mover (tensor_load_to_lds + s_wait_tensorcnt) with
// per-row LDS padding. Round-3: branchless pointer-stepped GEMM epilogue,
// vectorized LayerNorm. Workspace: ~352 MB.
// ---------------------------------------------------------------------------

#define D_MODEL 1024
#define N_HEADS 16
#define D_HEAD  64
#define SEQ     1024
#define BATCH   8
#define D_FF    4096
#define N_LAYERS 2

typedef __bf16 bf16;
typedef unsigned int       u32;
typedef unsigned long long u64;
typedef __attribute__((ext_vector_type(4)))  __bf16 bf16x4;
typedef __attribute__((ext_vector_type(8)))  __bf16 bf16x8;
typedef __attribute__((ext_vector_type(16))) __bf16 bf16x16;
typedef __attribute__((ext_vector_type(8)))  float  floatx8;
typedef __attribute__((ext_vector_type(4)))  u32    u32x4;
typedef __attribute__((ext_vector_type(4)))  int    i32x4;
typedef __attribute__((ext_vector_type(8)))  int    i32x8;

__device__ __forceinline__ floatx8 zero8() {
  floatx8 z = {0.f,0.f,0.f,0.f,0.f,0.f,0.f,0.f};
  return z;
}

__device__ __forceinline__ bf16x4 pack4(float4 v) {
  bf16x4 o;
  o[0] = (bf16)v.x; o[1] = (bf16)v.y; o[2] = (bf16)v.z; o[3] = (bf16)v.w;
  return o;
}

// A/B fragment per CDNA5 ISA 16-bit 16x32 layout:
// lane half hl: elements 0..7 = row[hl*8 .. hl*8+7], 8..15 = row[16+hl*8 ..]
__device__ __forceinline__ bf16x16 ldsfrag(const bf16* rowbase, int hl) {
  bf16x8 lo = *(const bf16x8*)(rowbase + hl * 8);
  bf16x8 hi = *(const bf16x8*)(rowbase + 16 + hl * 8);
  bf16x16 f;
#pragma unroll
  for (int i = 0; i < 8; ++i) { f[i] = lo[i]; f[i + 8] = hi[i]; }
  return f;
}

__device__ __forceinline__ floatx8 wmma_bf16(bf16x16 a, bf16x16 b, floatx8 c) {
  return __builtin_amdgcn_wmma_f32_16x16x32_bf16(false, a, false, b,
                                                 (short)0, c, false, false);
}

__device__ __forceinline__ u32 lds_addr32(const void* p) {
  return (u32)(__UINTPTR_TYPE__)p;   // low 32 bits of generic ptr = LDS offset
}

// ---------------------------------------------------------------------------
// Tensor Data Mover: 2-D bf16 tile (tile_d0 x tile_d1 rows) global -> LDS,
// pad_amount DWORDs inserted after each pad_interval DWORDs (lands rows at the
// padded LDS strides the fragments expect).
//   pad_interval code: 3 -> 16 DW (64B rows),  4 -> 32 DW (128B rows)
//   pad_amount   code: 3 -> 4 DW (16B pad)
// ---------------------------------------------------------------------------
__device__ __forceinline__ void tdm_load_2d(
    const bf16* gsrc, u32 lds_byte, u32 tile_d0, u32 tile_d1,
    u64 row_stride_elems, u32 pad_interval_code, u32 pad_amount_code)
{
  const u64 ga = (u64)(__UINTPTR_TYPE__)gsrc;
  u32x4 g0;
  g0.x = 1u;                                   // count=1 (valid user D#)
  g0.y = lds_byte;                             // lds_addr
  g0.z = (u32)ga;                              // global_addr[31:0]
  g0.w = (u32)(ga >> 32) | (2u << 30);         // global_addr[56:32] | type=2
  const u32 td0 = 1u << 30;                    // tensor_dim0 (OOB never hit)
  const u32 td1 = 1u << 20;                    // tensor_dim1
  i32x8 g1;
  g1[0] = (int)((1u << 16) | (1u << 20) |
                (pad_interval_code << 22) | (pad_amount_code << 25));
  g1[1] = (int)((td0 & 0xFFFFu) << 16);
  g1[2] = (int)((td0 >> 16) | ((td1 & 0xFFFFu) << 16));
  g1[3] = (int)((td1 >> 16) | (tile_d0 << 16));
  g1[4] = (int)tile_d1;                        // tile_dim1 (tile_dim2 = 0)
  g1[5] = (int)(u32)(row_stride_elems & 0xFFFFFFFFull);
  g1[6] = (int)(u32)(row_stride_elems >> 32);
  g1[7] = 0;
  i32x4 z4 = {0, 0, 0, 0};
#if defined(__clang_major__) && (__clang_major__ >= 23)
  i32x8 z8 = {0, 0, 0, 0, 0, 0, 0, 0};
  __builtin_amdgcn_tensor_load_to_lds(g0, g1, z4, z4, z8, 0);
#else
  __builtin_amdgcn_tensor_load_to_lds(g0, g1, z4, z4, 0);
#endif
}

// ---------------------------------------------------------------------------
// GEMM: C[M,N] = A[M,K] @ Wt[N,K]^T + bias[N]
// A, Wt bf16; out fp32 or bf16.  mode: bit0 = relu, bit1 = bf16 output.
// 256 threads = 8 waves; tile 128x128, k-step 32; TDM-fed double buffers.
// ---------------------------------------------------------------------------
#define BM 128
#define BN 128
#define BK 32
#define LDT 40           // BK + 8 pad elems (80B rows; matches TDM pad 16B/64B)
#define TILE_A (BM * LDT)
#define TILE_B (BN * LDT)

__global__ __launch_bounds__(256) void gemm_bf16_kernel(
    const bf16* __restrict__ A, const bf16* __restrict__ Wt,
    const float* __restrict__ bias,
    float* __restrict__ Cf, bf16* __restrict__ Cb,
    int M, int N, int K, int mode)
{
  __shared__ __align__(16) bf16 AsAll[2 * TILE_A];
  __shared__ __align__(16) bf16 BsAll[2 * TILE_B];
  const int tid = threadIdx.x;
  const int wid = tid >> 5, lane = tid & 31;
  const int wm = wid & 3, wn = wid >> 2;
  const int hl = lane >> 4, l16 = lane & 15;
  const int bm0 = blockIdx.y * BM, bn0 = blockIdx.x * BN;

  floatx8 acc[2][4];
#pragma unroll
  for (int i = 0; i < 2; ++i)
#pragma unroll
    for (int j = 0; j < 4; ++j) acc[i][j] = zero8();

  const bf16* Abase = A + (size_t)bm0 * K;
  const bf16* Bbase = Wt + (size_t)bn0 * K;

  if (wid == 0) {   // prologue: fill buffer 0 via TDM
    tdm_load_2d(Abase, lds_addr32(&AsAll[0]), BK, BM, (u64)K, 3, 3);
    tdm_load_2d(Bbase, lds_addr32(&BsAll[0]), BK, BN, (u64)K, 3, 3);
    __builtin_amdgcn_s_wait_tensorcnt(0);
  }
  __syncthreads();

  const int nk = K / BK;
  for (int kt = 0; kt < nk; ++kt) {
    const int cur = kt & 1;
    if (wid == 0 && kt + 1 < nk) {   // stream next tiles into the other buffer
      const int k0n = (kt + 1) * BK;
      tdm_load_2d(Abase + k0n, lds_addr32(&AsAll[(cur ^ 1) * TILE_A]),
                  BK, BM, (u64)K, 3, 3);
      tdm_load_2d(Bbase + k0n, lds_addr32(&BsAll[(cur ^ 1) * TILE_B]),
                  BK, BN, (u64)K, 3, 3);
    }

    const bf16* Ac = &AsAll[cur * TILE_A];
    const bf16* Bc = &BsAll[cur * TILE_B];
    bf16x16 af[2], bfr[4];
#pragma unroll
    for (int tm = 0; tm < 2; ++tm)
      af[tm] = ldsfrag(&Ac[(wm * 32 + tm * 16 + l16) * LDT], hl);
#pragma unroll
    for (int tn = 0; tn < 4; ++tn)
      bfr[tn] = ldsfrag(&Bc[(wn * 64 + tn * 16 + l16) * LDT], hl);
#pragma unroll
    for (int tm = 0; tm < 2; ++tm)
#pragma unroll
      for (int tn = 0; tn < 4; ++tn)
        acc[tm][tn] = wmma_bf16(af[tm], bfr[tn], acc[tm][tn]);

    if (wid == 0 && kt + 1 < nk)
      __builtin_amdgcn_s_wait_tensorcnt(0);   // next buffer landed
    __syncthreads();
  }

  // ---- epilogue: branch-hoisted, pointer-stepped stores ----
  const int colbase = bn0 + wn * 64 + l16;
  const float clampLo = (mode & 1) ? 0.0f : -3.4e38f;   // relu via one fmax
  float bv[4];
#pragma unroll
  for (int tn = 0; tn < 4; ++tn) bv[tn] = bias[colbase + tn * 16];

  if (mode & 2) {   // bf16 output
#pragma unroll
    for (int tm = 0; tm < 2; ++tm) {
      const int row0 = bm0 + wm * 32 + tm * 16 + hl * 8;
      bf16* p0 = Cb + (size_t)row0 * N + colbase;
#pragma unroll
      for (int tn = 0; tn < 4; ++tn) {
        bf16* p = p0 + tn * 16;
#pragma unroll
        for (int v = 0; v < 8; ++v) {
          *p = (bf16)fmaxf(acc[tm][tn][v] + bv[tn], clampLo);
          p += N;
        }
      }
    }
  } else {          // fp32 output
#pragma unroll
    for (int tm = 0; tm < 2; ++tm) {
      const int row0 = bm0 + wm * 32 + tm * 16 + hl * 8;
      float* p0 = Cf + (size_t)row0 * N + colbase;
#pragma unroll
      for (int tn = 0; tn < 4; ++tn) {
        float* p = p0 + tn * 16;
#pragma unroll
        for (int v = 0; v < 8; ++v) {
          *p = fmaxf(acc[tm][tn][v] + bv[tn], clampLo);
          p += N;
        }
      }
    }
  }
}

// ---------------------------------------------------------------------------
// Flash attention (causal), all-bf16 operands. One block per (b,h,128-q-chunk).
// Q/K tiles via TDM; V micro-transposed in registers; scale applied post-WMMA.
// ---------------------------------------------------------------------------
__global__ __launch_bounds__(256) void flash_attn_kernel(
    const bf16* __restrict__ Qg, const bf16* __restrict__ Kg,
    const bf16* __restrict__ Vg, bf16* __restrict__ Og)
{
  // sh1: Qs[128][72] (prologue only) then Vt[64][136]  (max 9216 elems)
  // sh2: Ks[128][72] then Ps[8][16][136]               (max 17408 elems)
  __shared__ __align__(16) bf16 sh1[9216];
  __shared__ __align__(16) bf16 sh2[17408];

  const int tid = threadIdx.x;
  const int wid = tid >> 5, lane = tid & 31;
  const int hl = lane >> 4, l16 = lane & 15;
  const int qc = blockIdx.x;
  const int bh = blockIdx.y;
  const int b = bh >> 4;
  const int h = bh & 15;
  const float scale = 0.125f;  // 1/sqrt(64)

  const bf16* Qb = Qg + ((size_t)b * SEQ + (size_t)qc * 128) * D_MODEL + h * D_HEAD;
  const bf16* Kb = Kg + (size_t)b * SEQ * D_MODEL + h * D_HEAD;
  const bf16* Vb = Vg + (size_t)b * SEQ * D_MODEL + h * D_HEAD;

  // ---- Q chunk -> sh1 via TDM (raw copy; 128B rows + 16B pad -> stride 72) --
  if (wid == 0) {
    tdm_load_2d(Qb, lds_addr32(sh1), D_HEAD, 128, (u64)D_MODEL, 4, 3);
    __builtin_amdgcn_s_wait_tensorcnt(0);
  }
  __syncthreads();
  bf16x16 qf[2];
#pragma unroll
  for (int kk = 0; kk < 2; ++kk)
    qf[kk] = ldsfrag(&sh1[(wid * 16 + l16) * 72 + kk * 32], hl);

  floatx8 oacc[4];
#pragma unroll
  for (int i = 0; i < 4; ++i) oacc[i] = zero8();
  float mrun[8], lrun[8];
#pragma unroll
  for (int v = 0; v < 8; ++v) { mrun[v] = -1e30f; lrun[v] = 0.f; }

  const int nkb = qc + 1;
  for (int kb = 0; kb < nkb; ++kb) {
    __syncthreads();  // prev iteration's Ps/Vt reads done
    // ---- K block -> sh2 via TDM ----
    if (wid == 0)
      tdm_load_2d(Kb + (size_t)kb * 128 * D_MODEL, lds_addr32(sh2),
                  D_HEAD, 128, (u64)D_MODEL, 4, 3);
    // ---- V block -> sh1 transposed Vt[d][key] via 4x4 micro-transpose ----
    {
      const bf16* Vbk = Vb + (size_t)kb * 128 * D_MODEL;
      bf16x4 vr[2][4];
#pragma unroll
      for (int bb = 0; bb < 2; ++bb) {
        int blk = tid + bb * 256;       // 0..511 : 32 r-groups x 16 c-groups
        int cb = (blk & 15) << 2;
        int rb = (blk >> 4) << 2;
#pragma unroll
        for (int j = 0; j < 4; ++j)
          vr[bb][j] = *(const bf16x4*)(Vbk + (size_t)(rb + j) * D_MODEL + cb);
      }
#pragma unroll
      for (int bb = 0; bb < 2; ++bb) {
        int blk = tid + bb * 256;
        int cb = (blk & 15) << 2;
        int rb = (blk >> 4) << 2;
#pragma unroll
        for (int c = 0; c < 4; ++c) {
          bf16x4 t;
          t[0] = vr[bb][0][c]; t[1] = vr[bb][1][c];
          t[2] = vr[bb][2][c]; t[3] = vr[bb][3][c];
          *(bf16x4*)&sh1[(cb + c) * 136 + rb] = t;
        }
      }
    }
    if (wid == 0) __builtin_amdgcn_s_wait_tensorcnt(0);
    __syncthreads();

    // ---- S = Q @ K^T (raw), then scale ----
    floatx8 s[8];
#pragma unroll
    for (int tn = 0; tn < 8; ++tn) {
      s[tn] = zero8();
#pragma unroll
      for (int kk = 0; kk < 2; ++kk)
        s[tn] = wmma_bf16(qf[kk],
                          ldsfrag(&sh2[(tn * 16 + l16) * 72 + kk * 32], hl),
                          s[tn]);
    }
#pragma unroll
    for (int tn = 0; tn < 8; ++tn)
#pragma unroll
      for (int v = 0; v < 8; ++v) s[tn][v] *= scale;
    if (kb == qc) {  // causal mask on diagonal block
#pragma unroll
      for (int tn = 0; tn < 8; ++tn) {
        int colL = tn * 16 + l16;
#pragma unroll
        for (int v = 0; v < 8; ++v) {
          int rowL = wid * 16 + hl * 8 + v;
          if (colL > rowL) s[tn][v] = -1e30f;
        }
      }
    }
    // ---- online softmax (wave owns whole rows; shfl within 16 lanes) ----
    float mnew[8], alpha[8];
#pragma unroll
    for (int v = 0; v < 8; ++v) {
      float mx = s[0][v];
#pragma unroll
      for (int tn = 1; tn < 8; ++tn) mx = fmaxf(mx, s[tn][v]);
      mx = fmaxf(mx, __shfl_xor(mx, 1, 32));
      mx = fmaxf(mx, __shfl_xor(mx, 2, 32));
      mx = fmaxf(mx, __shfl_xor(mx, 4, 32));
      mx = fmaxf(mx, __shfl_xor(mx, 8, 32));
      mnew[v] = fmaxf(mrun[v], mx);
      alpha[v] = __expf(mrun[v] - mnew[v]);
      mrun[v] = mnew[v];
    }
    __syncthreads();  // everyone done reading Ks before Ps overwrites sh2

    float rs[8];
#pragma unroll
    for (int v = 0; v < 8; ++v) rs[v] = 0.f;
#pragma unroll
    for (int tn = 0; tn < 8; ++tn)
#pragma unroll
      for (int v = 0; v < 8; ++v) {
        float p = __expf(s[tn][v] - mnew[v]);
        rs[v] += p;
        sh2[wid * 2176 + (hl * 8 + v) * 136 + tn * 16 + l16] = (bf16)p;
      }
#pragma unroll
    for (int v = 0; v < 8; ++v) {
      float r = rs[v];
      r += __shfl_xor(r, 1, 32);
      r += __shfl_xor(r, 2, 32);
      r += __shfl_xor(r, 4, 32);
      r += __shfl_xor(r, 8, 32);
      lrun[v] = lrun[v] * alpha[v] + r;
    }
#pragma unroll
    for (int dt = 0; dt < 4; ++dt)
#pragma unroll
      for (int v = 0; v < 8; ++v) oacc[dt][v] *= alpha[v];

    // wave-local: our P stores must land before our fragment reads of them
    asm volatile("s_wait_dscnt 0x0" ::: "memory");

    // ---- O += P @ V  (K-dim = 128 keys, 4 k-steps) ----
#pragma unroll
    for (int dt = 0; dt < 4; ++dt)
#pragma unroll
      for (int ks = 0; ks < 4; ++ks) {
        bf16x16 pa = ldsfrag(&sh2[wid * 2176 + l16 * 136 + ks * 32], hl);
        bf16x16 vb = ldsfrag(&sh1[(dt * 16 + l16) * 136 + ks * 32], hl);
        oacc[dt] = wmma_bf16(pa, vb, oacc[dt]);
      }
  }

  bf16* Ob = Og + ((size_t)b * SEQ + (size_t)qc * 128) * D_MODEL + h * D_HEAD;
#pragma unroll
  for (int v = 0; v < 8; ++v) {
    float inv = 1.0f / lrun[v];
    int rowL = wid * 16 + hl * 8 + v;
#pragma unroll
    for (int dt = 0; dt < 4; ++dt) {
      int col = dt * 16 + l16;
      Ob[(size_t)rowL * D_MODEL + col] = (bf16)(oacc[dt][v] * inv);
    }
  }
}

// ---------------------------------------------------------------------------
// out = LayerNorm(a + b) * g + beta, dual-written fp32 + bf16 (GEMM feed).
// One block per row; each thread owns 4 consecutive columns (vectorized).
// ---------------------------------------------------------------------------
__global__ __launch_bounds__(256) void add_ln_kernel(
    const float* __restrict__ a, const float* __restrict__ b,
    const float* __restrict__ g, const float* __restrict__ bet,
    float* __restrict__ outf, bf16* __restrict__ outb)
{
  __shared__ float sm1[8], sm2[8];
  const int tid = threadIdx.x;
  const size_t base = (size_t)blockIdx.x * D_MODEL;
  const int c0 = tid * 4;
  float4 av = *(const float4*)(a + base + c0);
  float4 bv = *(const float4*)(b + base + c0);
  float x[4] = {av.x + bv.x, av.y + bv.y, av.z + bv.z, av.w + bv.w};
  float s = x[0] + x[1] + x[2] + x[3];
#pragma unroll
  for (int m = 16; m >= 1; m >>= 1) s += __shfl_xor(s, m, 32);
  if ((tid & 31) == 0) sm1[tid >> 5] = s;
  __syncthreads();
  float tot = 0.f;
#pragma unroll
  for (int i = 0; i < 8; ++i) tot += sm1[i];
  float mean = tot * (1.0f / D_MODEL);
  float s2 = 0.f;
#pragma unroll
  for (int i = 0; i < 4; ++i) { float d = x[i] - mean; s2 += d * d; }
#pragma unroll
  for (int m = 16; m >= 1; m >>= 1) s2 += __shfl_xor(s2, m, 32);
  if ((tid & 31) == 0) sm2[tid >> 5] = s2;
  __syncthreads();
  float tv = 0.f;
#pragma unroll
  for (int i = 0; i < 8; ++i) tv += sm2[i];
  float rstd = rsqrtf(tv * (1.0f / D_MODEL) + 1e-6f);
  float4 gv = *(const float4*)(g + c0);
  float4 ev = *(const float4*)(bet + c0);
  float4 y;
  y.x = (x[0] - mean) * rstd * gv.x + ev.x;
  y.y = (x[1] - mean) * rstd * gv.y + ev.y;
  y.z = (x[2] - mean) * rstd * gv.z + ev.z;
  y.w = (x[3] - mean) * rstd * gv.w + ev.w;
  *(float4*)(outf + base + c0) = y;
  *(bf16x4*)(outb + base + c0) = pack4(y);
}

// ---------------------------------------------------------------------------
// One-time prep: fp32 -> bf16 elementwise; fp32[K][N] -> bf16[N][K] transpose.
// ---------------------------------------------------------------------------
__global__ __launch_bounds__(256) void cvt_bf16_kernel(
    const float* __restrict__ in, bf16* __restrict__ outp)
{
  size_t i = ((size_t)blockIdx.x * 256 + threadIdx.x) * 4;
  float4 v = *(const float4*)(in + i);
  *(bf16x4*)(outp + i) = pack4(v);
}

__global__ __launch_bounds__(256) void transpose_w_kernel(
    const float* __restrict__ src, bf16* __restrict__ dst, int K, int N)
{
  __shared__ __align__(16) bf16 t[32 * 36];
  const int tid = threadIdx.x;
  const int bx = blockIdx.x << 5;  // n
  const int by = blockIdx.y << 5;  // k
  {
    int r = tid >> 3;            // k-local 0..31
    int c = (tid & 7) << 2;      // n-local 0,4,..,28
    float4 v = *(const float4*)(src + (size_t)(by + r) * N + bx + c);
    t[(c + 0) * 36 + r] = (bf16)v.x;
    t[(c + 1) * 36 + r] = (bf16)v.y;
    t[(c + 2) * 36 + r] = (bf16)v.z;
    t[(c + 3) * 36 + r] = (bf16)v.w;
  }
  __syncthreads();
  {
    int rn = tid >> 3;           // n-local
    int ck = (tid & 7) << 2;     // k-local
    bf16x4 o;
    o[0] = t[rn * 36 + ck + 0];
    o[1] = t[rn * 36 + ck + 1];
    o[2] = t[rn * 36 + ck + 2];
    o[3] = t[rn * 36 + ck + 3];
    *(bf16x4*)&dst[(size_t)(bx + rn) * K + by + ck] = o;
  }
}

// ---------------------------------------------------------------------------
extern "C" void kernel_launch(void* const* d_in, const int* in_sizes, int n_in,
                              void* d_out, int out_size, void* d_ws, size_t ws_size,
                              hipStream_t stream) {
  (void)in_sizes; (void)n_in; (void)out_size; (void)ws_size;
  const float* x_in = (const float*)d_in[0];
  // d_in[1] = mask (causality computed analytically in the flash kernel)
  const float* Wq = (const float*)d_in[2];
  const float* bq = (const float*)d_in[3];
  const float* Wk = (const float*)d_in[4];
  const float* bk = (const float*)d_in[5];
  const float* Wv = (const float*)d_in[6];
  const float* bv = (const float*)d_in[7];
  const float* Wo = (const float*)d_in[8];
  const float* bo = (const float*)d_in[9];
  const float* W1 = (const float*)d_in[10];
  const float* b1 = (const float*)d_in[11];
  const float* W2 = (const float*)d_in[12];
  const float* b2 = (const float*)d_in[13];
  const float* g1 = (const float*)d_in[14];
  const float* be1 = (const float*)d_in[15];
  const float* g2 = (const float*)d_in[16];
  const float* be2 = (const float*)d_in[17];
  float* out = (float*)d_out;

  const int M = BATCH * SEQ;                // 8192
  const size_t DD = (size_t)D_MODEL * D_MODEL;
  const size_t DF = (size_t)D_MODEL * D_FF;
  const size_t NA = (size_t)M * D_MODEL;    // per-activation element count

  // ---- workspace carve-up (bf16 region, then fp32 region) ----
  bf16* wsb = (bf16*)d_ws;
  size_t off = 0;
  bf16* wtq = wsb + off; off += N_LAYERS * DD;
  bf16* wtk = wsb + off; off += N_LAYERS * DD;
  bf16* wtv = wsb + off; off += N_LAYERS * DD;
  bf16* wto = wsb + off; off += N_LAYERS * DD;
  bf16* wt1 = wsb + off; off += N_LAYERS * DF;   // [DFF][D]
  bf16* wt2 = wsb + off; off += N_LAYERS * DF;   // [D][DFF]
  bf16* xb  = wsb + off; off += NA;
  bf16* qb_ = wsb + off; off += NA;
  bf16* kb_ = wsb + off; off += NA;
  bf16* vb_ = wsb + off; off += NA;
  bf16* ab_ = wsb + off; off += NA;
  bf16* x1b = wsb + off; off += NA;
  bf16* x2b = wsb + off; off += NA;
  bf16* hb_ = wsb + off; off += (size_t)M * D_FF;
  float* wsf = (float*)(wsb + off);
  float* bo_f = wsf;
  float* x1f  = wsf + NA;
  float* bf_f = wsf + 2 * NA;
  float* x2f  = wsf + 3 * NA;

  dim3 blk(256);
  dim3 gD(D_MODEL / BN, M / BM);   // (8, 64)
  dim3 gF1(D_FF / BN, M / BM);     // (32, 64)
  dim3 gAttn(SEQ / 128, BATCH * N_HEADS);
  dim3 gLN(M);
  dim3 gTd(D_MODEL / 32, D_MODEL / 32);
  dim3 gT1(D_FF / 32, D_MODEL / 32);   // W1[K=D][N=DFF] -> [DFF][D]
  dim3 gT2(D_MODEL / 32, D_FF / 32);   // W2[K=DFF][N=D] -> [D][DFF]

  // ---- one-time prep: bf16 weights (transposed) + bf16 x ----
  for (int l = 0; l < N_LAYERS; ++l) {
    transpose_w_kernel<<<gTd, blk, 0, stream>>>(Wq + l * DD, wtq + l * DD, D_MODEL, D_MODEL);
    transpose_w_kernel<<<gTd, blk, 0, stream>>>(Wk + l * DD, wtk + l * DD, D_MODEL, D_MODEL);
    transpose_w_kernel<<<gTd, blk, 0, stream>>>(Wv + l * DD, wtv + l * DD, D_MODEL, D_MODEL);
    transpose_w_kernel<<<gTd, blk, 0, stream>>>(Wo + l * DD, wto + l * DD, D_MODEL, D_MODEL);
    transpose_w_kernel<<<gT1, blk, 0, stream>>>(W1 + l * DF, wt1 + l * DF, D_MODEL, D_FF);
    transpose_w_kernel<<<gT2, blk, 0, stream>>>(W2 + l * DF, wt2 + l * DF, D_FF, D_MODEL);
  }
  cvt_bf16_kernel<<<(unsigned)(NA / 1024), blk, 0, stream>>>(x_in, xb);

  for (int l = 0; l < N_LAYERS; ++l) {
    const bf16* xcb = (l == 0) ? xb : x2b;
    const float* xcf = (l == 0) ? x_in : x2f;
    gemm_bf16_kernel<<<gD, blk, 0, stream>>>(xcb, wtq + l * DD, bq + (size_t)l * D_MODEL,
                                             bo_f, qb_, M, D_MODEL, D_MODEL, 2);
    gemm_bf16_kernel<<<gD, blk, 0, stream>>>(xcb, wtk + l * DD, bk + (size_t)l * D_MODEL,
                                             bo_f, kb_, M, D_MODEL, D_MODEL, 2);
    gemm_bf16_kernel<<<gD, blk, 0, stream>>>(xcb, wtv + l * DD, bv + (size_t)l * D_MODEL,
                                             bo_f, vb_, M, D_MODEL, D_MODEL, 2);
    flash_attn_kernel<<<gAttn, blk, 0, stream>>>(qb_, kb_, vb_, ab_);
    gemm_bf16_kernel<<<gD, blk, 0, stream>>>(ab_, wto + l * DD, bo + (size_t)l * D_MODEL,
                                             bo_f, qb_, M, D_MODEL, D_MODEL, 0);
    add_ln_kernel<<<gLN, blk, 0, stream>>>(bo_f, xcf, g1 + (size_t)l * D_MODEL,
                                           be1 + (size_t)l * D_MODEL, x1f, x1b);
    gemm_bf16_kernel<<<gF1, blk, 0, stream>>>(x1b, wt1 + l * DF, b1 + (size_t)l * D_FF,
                                              bf_f, hb_, M, D_FF, D_MODEL, 3);
    gemm_bf16_kernel<<<gD, blk, 0, stream>>>(hb_, wt2 + l * DF, b2 + (size_t)l * D_MODEL,
                                             bf_f, hb_, M, D_MODEL, D_FF, 0);
    add_ln_kernel<<<gLN, blk, 0, stream>>>(x1f, bf_f, g2 + (size_t)l * D_MODEL,
                                           be2 + (size_t)l * D_MODEL,
                                           (l == N_LAYERS - 1) ? out : x2f, x2b);
  }
}
